// JumpODEEncoder_73770358276745
// MI455X (gfx1250) — matrix operand order repaired and checked by reading, not compiled, not run on hardware
//
#include <hip/hip_runtime.h>
#include <math.h>
#include <stdint.h>

// ---------------------------------------------------------------------------
// Model dims (fixed by the reference)
// ---------------------------------------------------------------------------
#define DM    512
#define TT    1024
#define BB    8
#define BT    (BB*TT)          // 8192 rows
#define MD    (BB*(TT-1))      // 8184 rows (RK4 stage tensors)
#define NH    8
#define HD    64

typedef __attribute__((ext_vector_type(16))) __bf16 bf16x16;
typedef __attribute__((ext_vector_type(8)))  __bf16 bf16x8;
typedef __attribute__((ext_vector_type(8)))  float  f32x8;

#define ACT_NONE 0
#define ACT_TANH 1
#define ACT_RELU 2

__device__ __forceinline__ f32x8 wmma_bf16(bf16x16 a, bf16x16 b, f32x8 c) {
  // v_wmma_f32_16x16x32_bf16  (neg_a, A, neg_b, B, c_mod, C, reuse_a, reuse_b)
  return __builtin_amdgcn_wmma_f32_16x16x32_bf16(false, a, false, b, (short)0, c,
                                                 false, false);
}

__device__ __forceinline__ float sigmoidf_(float x) {
  return 1.0f / (1.0f + __expf(-x));
}

// A-fragment (16x32 bf16, row striped per lane): lane(0..15)->M row, half
// selects K sub-blocks {half*8..half*8+7, 16+half*8..}
__device__ __forceinline__ bf16x16 load_a_frag(const __bf16* __restrict__ A,
                                               long row, int ldk, int k0,
                                               int half) {
  bf16x16 a;
  const __bf16* p = A + row * (long)ldk + k0 + half * 8;
  bf16x8 lo = *(const bf16x8*)p;
  bf16x8 hi = *(const bf16x8*)(p + 16);
#pragma unroll
  for (int i = 0; i < 8; ++i) { a[i] = lo[i]; a[8 + i] = hi[i]; }
  return a;
}

// B-fragment (32x16 bf16, col per lane): lane(0..15)->N col, half selects
// K rows {half*8..half*8+7, 16+half*8..}; W stored (N,K) row-major.
__device__ __forceinline__ bf16x16 load_b_frag(const __bf16* __restrict__ W,
                                               long col, int ldk, int k0,
                                               int half) {
  const __bf16* p = W + col * (long)ldk + k0 + half * 8;
  bf16x8 lo = *(const bf16x8*)p;
  bf16x8 hi = *(const bf16x8*)(p + 16);
  bf16x16 b;
#pragma unroll
  for (int i = 0; i < 8; ++i) { b[i] = lo[i]; b[8 + i] = hi[i]; }
  return b;
}

// ---------------------------------------------------------------------------
// f32 -> bf16 cast
// ---------------------------------------------------------------------------
__global__ void k_cast(const float* __restrict__ in, __bf16* __restrict__ out,
                       int n) {
  int i = blockIdx.x * 256 + threadIdx.x;
  if (i < n) out[i] = (__bf16)in[i];
}

// ---------------------------------------------------------------------------
// Generic bf16 WMMA GEMM:  C[M,N] = act(A[M,K] @ W[N,K]^T + bias) (+ add)
// One 16x64 output strip per wave (4 N-tiles share one A fragment),
// 8 waves per block. A-row is clamped (branch-free inner loop); ragged M
// rows only masked at the store. K-loop is software-pipelined (double-
// buffered fragments) so VMEM latency overlaps the WMMA issue.
// ---------------------------------------------------------------------------
__global__ __launch_bounds__(256) void k_gemm(
    const __bf16* __restrict__ A, const __bf16* __restrict__ W,
    const float* __restrict__ bias, const float* __restrict__ add,
    float* __restrict__ Cf, __bf16* __restrict__ Cb, int M, int N, int K,
    int ldc, int act) {
  int lane = threadIdx.x & 31;
  int wave = threadIdx.x >> 5;
  int tilesN64 = N >> 6;
  int tilesM = (M + 15) >> 4;
  int strip = blockIdx.x * 8 + wave;
  if (strip >= tilesM * tilesN64) return;
  int tm = strip / tilesN64, tn = strip - tm * tilesN64;
  int l15 = lane & 15, half = lane >> 4;
  long arow = tm * 16 + l15;
  if (arow >= M) arow = M - 1;               // clamp: branch-free inner loop
  long bcol0 = tn * 64 + l15;

  f32x8 acc[4];
#pragma unroll
  for (int nt = 0; nt < 4; ++nt)
#pragma unroll
    for (int r = 0; r < 8; ++r) acc[nt][r] = 0.0f;

  // prologue: fragments for k0 = 0
  bf16x16 a = load_a_frag(A, arow, K, 0, half);
  bf16x16 bfr[4];
#pragma unroll
  for (int nt = 0; nt < 4; ++nt)
    bfr[nt] = load_b_frag(W, bcol0 + nt * 16, K, 0, half);

  for (int k0 = 32; k0 < K; k0 += 32) {
    // issue next chunk's loads first, then WMMA on the previous chunk
    __builtin_prefetch((const void*)(A + arow * (long)K + k0 + 128), 0, 1);
    bf16x16 an = load_a_frag(A, arow, K, k0, half);
    bf16x16 bn[4];
#pragma unroll
    for (int nt = 0; nt < 4; ++nt)
      bn[nt] = load_b_frag(W, bcol0 + nt * 16, K, k0, half);
#pragma unroll
    for (int nt = 0; nt < 4; ++nt) acc[nt] = wmma_bf16(a, bfr[nt], acc[nt]);
    a = an;
#pragma unroll
    for (int nt = 0; nt < 4; ++nt) bfr[nt] = bn[nt];
  }
#pragma unroll
  for (int nt = 0; nt < 4; ++nt) acc[nt] = wmma_bf16(a, bfr[nt], acc[nt]);

#pragma unroll
  for (int nt = 0; nt < 4; ++nt) {
    int col = tn * 64 + nt * 16 + l15;
    float bv = bias ? bias[col] : 0.0f;
#pragma unroll
    for (int r = 0; r < 8; ++r) {
      int row = tm * 16 + r + 8 * half;
      if (row >= M) continue;
      float v = acc[nt][r] + bv;
      if (act == ACT_TANH) v = tanhf(v);
      else if (act == ACT_RELU) v = fmaxf(v, 0.0f);
      long off = (long)row * ldc + col;
      if (add) v += add[off];
      if (Cf) Cf[off] = v;
      if (Cb) Cb[off] = (__bf16)v;
    }
  }
}

// ---------------------------------------------------------------------------
// Flash-style attention (bf16 in, bf16 out, f32 online softmax).
// QKV packed rows of `rs` elements; grid = (qtilegroups, heads, batch).
// One wave handles one 16-row query tile; keys processed in 32-wide chunks.
// V chunks are staged into LDS via async global->LDS DMA (ASYNCcnt).
// ---------------------------------------------------------------------------
__global__ __launch_bounds__(256) void k_flash(
    const __bf16* __restrict__ QKV, int qo, int ko, int vo, int rs,
    __bf16* __restrict__ O, int os, int T, int causal) {
  __shared__ __bf16 Plds[8][16][32];   // 8 KB
  __shared__ __bf16 Vlds[8][32][64];   // 32 KB (per-wave 32x64 V chunk)
  int lane = threadIdx.x & 31;
  int wave = threadIdx.x >> 5;
  int b = blockIdx.z, h = blockIdx.y;
  int qt = blockIdx.x * 8 + wave;
  int qbase = qt * 16;
  int l15 = lane & 15, half = lane >> 4;

  const __bf16* Qp = QKV + (long)b * T * rs + qo + h * HD;
  const __bf16* Kp = QKV + (long)b * T * rs + ko + h * HD;
  const __bf16* Vp = QKV + (long)b * T * rs + vo + h * HD;

  // Q fragments for K-chunks d=0..31, d=32..63
  bf16x16 qa[2];
#pragma unroll
  for (int kc = 0; kc < 2; ++kc) {
    const __bf16* p = Qp + (long)(qbase + l15) * rs + kc * 32 + half * 16;
    bf16x8 lo = *(const bf16x8*)p;
    bf16x8 hi = *(const bf16x8*)(p + 8);
#pragma unroll
    for (int i = 0; i < 8; ++i) { qa[kc][i] = lo[i]; qa[kc][8 + i] = hi[i]; }
  }

  float m_i[8], l_i[8];
  f32x8 oc[4];
#pragma unroll
  for (int r = 0; r < 8; ++r) { m_i[r] = -1e30f; l_i[r] = 0.0f; }
#pragma unroll
  for (int d = 0; d < 4; ++d)
#pragma unroll
    for (int r = 0; r < 8; ++r) oc[d][r] = 0.0f;

  unsigned vbase = (unsigned)(uintptr_t)(&Vlds[wave][0][0]);

  const float sc = 0.125f;  // 1/sqrt(64)
  int jend = causal ? (qbase + 16) : T;
  for (int j0 = 0; j0 < jend; j0 += 32) {
    // ---- async-stage the 32x64 bf16 V chunk into LDS (8 x 512B) ----
    {
#pragma unroll
      for (int i = 0; i < 8; ++i) {
        int c = i * 32 + lane;              // 16-byte chunk id, 256 total
        int row = c >> 3, colc = (c & 7) * 8;
        const __bf16* g = Vp + (long)(j0 + row) * rs + colc;
        unsigned laddr = vbase + (unsigned)(row * 64 + colc) * 2u;
        asm volatile("global_load_async_to_lds_b128 %0, %1, off"
                     :: "v"(laddr), "v"(g) : "memory");
      }
    }
    // ---- S = Q @ K^T while the V DMA is in flight ----
    f32x8 s0 = {0.f, 0.f, 0.f, 0.f, 0.f, 0.f, 0.f, 0.f};
    f32x8 s1 = {0.f, 0.f, 0.f, 0.f, 0.f, 0.f, 0.f, 0.f};
#pragma unroll
    for (int kc = 0; kc < 2; ++kc) {
      bf16x16 kb0, kb1;
      {
        const __bf16* p = Kp + (long)(j0 + l15) * rs + kc * 32 + half * 16;
        bf16x8 lo = *(const bf16x8*)p;
        bf16x8 hi = *(const bf16x8*)(p + 8);
#pragma unroll
        for (int i = 0; i < 8; ++i) { kb0[i] = lo[i]; kb0[8 + i] = hi[i]; }
      }
      {
        const __bf16* p = Kp + (long)(j0 + 16 + l15) * rs + kc * 32 + half * 16;
        bf16x8 lo = *(const bf16x8*)p;
        bf16x8 hi = *(const bf16x8*)(p + 8);
#pragma unroll
        for (int i = 0; i < 8; ++i) { kb1[i] = lo[i]; kb1[8 + i] = hi[i]; }
      }
      s0 = wmma_bf16(qa[kc], kb0, s0);
      s1 = wmma_bf16(qa[kc], kb1, s1);
    }
    // ---- online softmax over the 32-key chunk ----
    float corr[8];
#pragma unroll
    for (int r = 0; r < 8; ++r) {
      float v0 = s0[r] * sc, v1 = s1[r] * sc;
      if (causal) {
        int q = qbase + r + 8 * half;
        if (j0 + l15 > q) v0 = -1e30f;
        if (j0 + 16 + l15 > q) v1 = -1e30f;
      }
      float m = fmaxf(v0, v1);
#pragma unroll
      for (int off = 1; off < 16; off <<= 1)
        m = fmaxf(m, __shfl_xor(m, off, 32));
      float mnew = fmaxf(m_i[r], m);
      corr[r] = __expf(m_i[r] - mnew);
      float p0 = __expf(v0 - mnew);
      float p1 = __expf(v1 - mnew);
      float ps = p0 + p1;
#pragma unroll
      for (int off = 1; off < 16; off <<= 1) ps += __shfl_xor(ps, off, 32);
      l_i[r] = l_i[r] * corr[r] + ps;
      m_i[r] = mnew;
      int row = r + 8 * half;
      Plds[wave][row][l15] = (__bf16)p0;
      Plds[wave][row][16 + l15] = (__bf16)p1;
    }
#pragma unroll
    for (int d = 0; d < 4; ++d)
#pragma unroll
      for (int r = 0; r < 8; ++r) oc[d][r] *= corr[r];

    // wave-local LDS ordering for P; async V copy completion
    asm volatile("s_wait_dscnt 0" ::: "memory");
    asm volatile("s_wait_asynccnt 0" ::: "memory");

    // P in A-fragment layout (K = 32 keys)
    bf16x16 pa;
#pragma unroll
    for (int i = 0; i < 8; ++i) {
      pa[i] = Plds[wave][l15][half * 16 + i];
      pa[8 + i] = Plds[wave][l15][half * 16 + 8 + i];
    }
    // O += P @ V  (V as B-fragments gathered from LDS, 4 tiles of 16 cols)
#pragma unroll
    for (int dt = 0; dt < 4; ++dt) {
      bf16x16 vb;
      int d = dt * 16 + l15;
#pragma unroll
      for (int i = 0; i < 16; ++i)
        vb[i] = Vlds[wave][half * 16 + i][d];
      oc[dt] = wmma_bf16(pa, vb, oc[dt]);
    }
  }
  float inv[8];
#pragma unroll
  for (int r = 0; r < 8; ++r) inv[r] = 1.0f / fmaxf(l_i[r], 1e-30f);
#pragma unroll
  for (int dt = 0; dt < 4; ++dt)
#pragma unroll
    for (int r = 0; r < 8; ++r) {
      int row = r + 8 * half;
      int q = qbase + row;
      O[(long)(b * TT + q) * os + h * HD + dt * 16 + l15] =
          (__bf16)(oc[dt][r] * inv[r]);
    }
}

// ---------------------------------------------------------------------------
// GRU scan: one workgroup, h(8x512) lives in LDS (bf16, rows 8..15 zero-pad),
// per step gh = h @ Whh^T + bhh via WMMA (32 waves x 3 N-tiles, A fragment
// hoisted, Whh B-fragments double-buffered over the K-loop).
// ---------------------------------------------------------------------------
__global__ __launch_bounds__(1024) void k_gru_scan(
    const float* __restrict__ gx, const __bf16* __restrict__ Whh,
    const float* __restrict__ bhh, float* __restrict__ Hs,
    __bf16* __restrict__ Hsb) {
  __shared__ __bf16 hb[16 * DM];        // 16 KB
  __shared__ __bf16 gh[8 * 3 * DM];     // 24 KB
  int tid = threadIdx.x;
  int lane = tid & 31, wave = tid >> 5;
  int l15 = lane & 15, half = lane >> 4;
  for (int i = tid; i < 16 * DM; i += 1024) hb[i] = (__bf16)0.0f;
  __syncthreads();

  long col[3];
#pragma unroll
  for (int nt = 0; nt < 3; ++nt) col[nt] = (wave + nt * 32) * 16 + l15;

  for (int t = 0; t < TT; ++t) {
    f32x8 c[3];
#pragma unroll
    for (int nt = 0; nt < 3; ++nt)
#pragma unroll
      for (int r = 0; r < 8; ++r) c[nt][r] = 0.0f;

    // prologue (k0 = 0)
    bf16x16 a;
    {
      const __bf16* ap = &hb[l15 * DM + half * 8];
#pragma unroll
      for (int i = 0; i < 8; ++i) { a[i] = ap[i]; a[8 + i] = ap[16 + i]; }
    }
    bf16x16 bfr[3];
#pragma unroll
    for (int nt = 0; nt < 3; ++nt) bfr[nt] = load_b_frag(Whh, col[nt], DM, 0, half);

    for (int k0 = 32; k0 < DM; k0 += 32) {
      bf16x16 an;
      {
        const __bf16* ap = &hb[l15 * DM + k0 + half * 8];
#pragma unroll
        for (int i = 0; i < 8; ++i) { an[i] = ap[i]; an[8 + i] = ap[16 + i]; }
      }
      bf16x16 bn[3];
#pragma unroll
      for (int nt = 0; nt < 3; ++nt) bn[nt] = load_b_frag(Whh, col[nt], DM, k0, half);
#pragma unroll
      for (int nt = 0; nt < 3; ++nt) c[nt] = wmma_bf16(a, bfr[nt], c[nt]);
      a = an;
#pragma unroll
      for (int nt = 0; nt < 3; ++nt) bfr[nt] = bn[nt];
    }
#pragma unroll
    for (int nt = 0; nt < 3; ++nt) c[nt] = wmma_bf16(a, bfr[nt], c[nt]);

#pragma unroll
    for (int nt = 0; nt < 3; ++nt) {
      float bv = bhh[col[nt]];
#pragma unroll
      for (int r = 0; r < 8; ++r) {
        int row = r + 8 * half;
        if (row < 8) gh[row * (3 * DM) + col[nt]] = (__bf16)(c[nt][r] + bv);
      }
    }
    __syncthreads();
    for (int u = tid; u < BB * DM; u += 1024) {
      int bb = u >> 9, j = u & (DM - 1);
      const float* g = gx + (long)(bb * TT + t) * (3 * DM);
      float rr = sigmoidf_(g[j] + (float)gh[bb * (3 * DM) + j]);
      float zz = sigmoidf_(g[DM + j] + (float)gh[bb * (3 * DM) + DM + j]);
      float hp = (float)hb[bb * DM + j];
      float nn = tanhf(g[2 * DM + j] + rr * (float)gh[bb * (3 * DM) + 2 * DM + j]);
      float hn = (1.0f - zz) * nn + zz * hp;
      hb[bb * DM + j] = (__bf16)hn;
      long o = (long)(bb * TT + t) * DM + j;
      Hs[o] = hn;
      Hsb[o] = (__bf16)hn;
    }
    __syncthreads();
  }
}

// ---------------------------------------------------------------------------
// LayerNorm over rows of 512 (block = 256 threads, 2 elems/thread)
// ---------------------------------------------------------------------------
__global__ __launch_bounds__(256) void k_ln(const float* __restrict__ in,
                                            const float* __restrict__ w,
                                            const float* __restrict__ bsh,
                                            float* __restrict__ outf,
                                            __bf16* __restrict__ outb) {
  __shared__ float red[16];
  int row = blockIdx.x, tid = threadIdx.x;
  const float* p = in + (long)row * DM;
  float v0 = p[tid], v1 = p[tid + 256];
  float s = v0 + v1, ss = v0 * v0 + v1 * v1;
#pragma unroll
  for (int off = 16; off >= 1; off >>= 1) {
    s += __shfl_xor(s, off, 32);
    ss += __shfl_xor(ss, off, 32);
  }
  int wv = tid >> 5;
  if ((tid & 31) == 0) { red[wv] = s; red[8 + wv] = ss; }
  __syncthreads();
  s = 0.f; ss = 0.f;
#pragma unroll
  for (int i = 0; i < 8; ++i) { s += red[i]; ss += red[8 + i]; }
  float mean = s * (1.0f / DM);
  float var = ss * (1.0f / DM) - mean * mean;
  float rstd = rsqrtf(var + 1e-5f);
  float o0 = (v0 - mean) * rstd * w[tid] + bsh[tid];
  float o1 = (v1 - mean) * rstd * w[tid + 256] + bsh[tid + 256];
  long base = (long)row * DM;
  if (outf) { outf[base + tid] = o0; outf[base + tid + 256] = o1; }
  if (outb) { outb[base + tid] = (__bf16)o0; outb[base + tid + 256] = (__bf16)o1; }
}

// ---------------------------------------------------------------------------
// RK4 helpers
// ---------------------------------------------------------------------------
__global__ void k_delta(const float* __restrict__ x, const float* __restrict__ ts,
                        float* __restrict__ df, __bf16* __restrict__ db) {
  long idx = (long)blockIdx.x * 256 + threadIdx.x;
  if (idx >= (long)MD * DM) return;
  int m = (int)(idx / DM), d = (int)(idx % DM);
  int b = m / (TT - 1), i = m % (TT - 1);
  float dt = ts[b * TT + i + 1] - ts[b * TT + i];
  float v = (x[(long)(b * TT + i + 1) * DM + d] - x[(long)(b * TT + i) * DM + d]) / dt;
  df[idx] = v;
  db[idx] = (__bf16)v;
}

__global__ void k_axpy_stage(const float* __restrict__ df,
                             const float* __restrict__ f,
                             const float* __restrict__ ts, float coef,
                             __bf16* __restrict__ outb) {
  long idx = (long)blockIdx.x * 256 + threadIdx.x;
  if (idx >= (long)MD * DM) return;
  int m = (int)(idx / DM);
  int b = m / (TT - 1), i = m % (TT - 1);
  float dt = ts[b * TT + i + 1] - ts[b * TT + i];
  outb[idx] = (__bf16)(df[idx] + coef * dt * f[idx]);
}

__global__ void k_rk4_final(const float* __restrict__ x,
                            const float* __restrict__ ts,
                            const float* __restrict__ f1,
                            const float* __restrict__ f2,
                            const float* __restrict__ f3,
                            const float* __restrict__ f4,
                            float* __restrict__ Hf, __bf16* __restrict__ Hb) {
  long idx = (long)blockIdx.x * 256 + threadIdx.x;
  if (idx >= (long)BT * DM) return;
  int bt = (int)(idx / DM), d = (int)(idx % DM);
  int t = bt % TT, b = bt / TT;
  float h = 0.0f;
  if (t > 0) {
    int i = t - 1;
    long m = (long)(b * (TT - 1) + i) * DM + d;
    float dt = ts[b * TT + i + 1] - ts[b * TT + i];
    h = x[(long)(b * TT + i) * DM + d] +
        dt * (1.0f / 6.0f) * (f1[m] + 2.0f * f2[m] + 2.0f * f3[m] + f4[m]);
  }
  Hf[idx] = h;
  Hb[idx] = (__bf16)h;
}

// ---------------------------------------------------------------------------
// Time-hybrid encoding: z = x + tm
// ---------------------------------------------------------------------------
__global__ void k_time_enc(const float* __restrict__ x,
                           const float* __restrict__ ts,
                           const float* __restrict__ te_a,
                           const float* __restrict__ te_b,
                           float* __restrict__ zf) {
  long idx = (long)blockIdx.x * 256 + threadIdx.x;
  if (idx >= (long)BT * DM) return;
  int bt = (int)(idx / DM), d = (int)(idx % DM);
  int t = bt % TT, b = bt / TT;
  float tsv = ts[b * TT + t];
  float tm;
  if (d & 1) {
    int i = (d - 1) >> 1;
    tm = tsv * te_a[i] + te_b[i];
  } else {
    int e = d >> 1;
    int f = e >> 1;
    float lf = (logf(2500.0f) / 127.0f) * (float)f;
    float freq = __expf(-lf) * (2.0f * 3.14159265358979f / 4.0f);
    float ph = tsv * freq;
    tm = (e & 1) ? __cosf(ph) : __sinf(ph);
  }
  zf[idx] = x[idx] + tm;
}

// ---------------------------------------------------------------------------
// Host orchestration
// ---------------------------------------------------------------------------
extern "C" void kernel_launch(void* const* d_in, const int* in_sizes, int n_in,
                              void* d_out, int out_size, void* d_ws,
                              size_t ws_size, hipStream_t stream) {
  (void)in_sizes; (void)n_in; (void)out_size; (void)ws_size;
  const float* x       = (const float*)d_in[0];
  const float* ts      = (const float*)d_in[1];
  const float* ode_w1  = (const float*)d_in[2];
  const float* ode_b1  = (const float*)d_in[3];
  const float* ode_w2  = (const float*)d_in[4];
  const float* ode_b2  = (const float*)d_in[5];
  const float* gru_wih = (const float*)d_in[6];
  const float* gru_whh = (const float*)d_in[7];
  const float* gru_bih = (const float*)d_in[8];
  const float* gru_bhh = (const float*)d_in[9];
  const float* te_a    = (const float*)d_in[10];
  const float* te_b    = (const float*)d_in[11];
  const float* t_in_w  = (const float*)d_in[12];
  const float* t_in_b  = (const float*)d_in[13];
  const float* t_out_w = (const float*)d_in[14];
  const float* t_out_b = (const float*)d_in[15];
  const float* t_ln1_w = (const float*)d_in[16];
  const float* t_ln1_b = (const float*)d_in[17];
  const float* t_ln2_w = (const float*)d_in[18];
  const float* t_ln2_b = (const float*)d_in[19];
  const float* t_ff1_w = (const float*)d_in[20];
  const float* t_ff1_b = (const float*)d_in[21];
  const float* t_ff2_w = (const float*)d_in[22];
  const float* t_ff2_b = (const float*)d_in[23];
  const float* t_lnf_w = (const float*)d_in[24];
  const float* t_lnf_b = (const float*)d_in[25];
  const float* ca_in_w = (const float*)d_in[26];
  const float* ca_in_b = (const float*)d_in[27];
  const float* ca_out_w= (const float*)d_in[28];
  const float* ca_out_b= (const float*)d_in[29];
  const float* ca_lin_w= (const float*)d_in[30];
  const float* ca_lin_b= (const float*)d_in[31];
  const float* ca_ln_w = (const float*)d_in[32];
  const float* ca_ln_b = (const float*)d_in[33];

  char* cur = (char*)d_ws;
  auto alloc = [&](size_t bytes) -> void* {
    void* r = (void*)cur;
    cur += ((bytes + 255) & ~(size_t)255);
    return r;
  };
  auto cast = [&](const float* src, __bf16* dst, int n) {
    k_cast<<<(n + 255) / 256, 256, 0, stream>>>(src, dst, n);
  };
  auto gemm = [&](const __bf16* A, const __bf16* W, const float* bias,
                  const float* add, float* Cf, __bf16* Cb, int M, int N, int K,
                  int ldc, int act) {
    int strips = ((M + 15) / 16) * (N / 64);   // N always a multiple of 64
    k_gemm<<<(strips + 7) / 8, 256, 0, stream>>>(A, W, bias, add, Cf, Cb, M, N,
                                                 K, ldc, act);
  };

  // bf16 weight copies
  __bf16* ode_w1b = (__bf16*)alloc(DM * DM * 2);
  __bf16* ode_w2b = (__bf16*)alloc(DM * DM * 2);
  __bf16* wihb    = (__bf16*)alloc(3 * DM * DM * 2);
  __bf16* whhb    = (__bf16*)alloc(3 * DM * DM * 2);
  __bf16* tinb    = (__bf16*)alloc(2 * 3 * DM * DM * 2);
  __bf16* toutb   = (__bf16*)alloc(2 * DM * DM * 2);
  __bf16* ff1b    = (__bf16*)alloc((size_t)2 * 4 * DM * DM * 2);
  __bf16* ff2b    = (__bf16*)alloc((size_t)2 * 4 * DM * DM * 2);
  __bf16* cainb   = (__bf16*)alloc(3 * DM * DM * 2);
  __bf16* caoutb  = (__bf16*)alloc(DM * DM * 2);
  __bf16* calinb  = (__bf16*)alloc(DM * DM * 2);
  cast(ode_w1, ode_w1b, DM * DM);
  cast(ode_w2, ode_w2b, DM * DM);
  cast(gru_wih, wihb, 3 * DM * DM);
  cast(gru_whh, whhb, 3 * DM * DM);
  cast(t_in_w, tinb, 2 * 3 * DM * DM);
  cast(t_out_w, toutb, 2 * DM * DM);
  cast(t_ff1_w, ff1b, 2 * 4 * DM * DM);
  cast(t_ff2_w, ff2b, 2 * 4 * DM * DM);
  cast(ca_in_w, cainb, 3 * DM * DM);
  cast(ca_out_w, caoutb, DM * DM);
  cast(ca_lin_w, calinb, DM * DM);

  // activations
  float*  deltaf = (float*)alloc((size_t)MD * DM * 4);
  __bf16* deltab = (__bf16*)alloc((size_t)MD * DM * 2);
  __bf16* stageb = (__bf16*)alloc((size_t)MD * DM * 2);
  __bf16* hidb   = (__bf16*)alloc((size_t)MD * DM * 2);
  float*  f1 = (float*)alloc((size_t)MD * DM * 4);
  float*  f2 = (float*)alloc((size_t)MD * DM * 4);
  float*  f3 = (float*)alloc((size_t)MD * DM * 4);
  float*  f4 = (float*)alloc((size_t)MD * DM * 4);
  float*  Hf = (float*)alloc((size_t)BT * DM * 4);
  __bf16* Hb = (__bf16*)alloc((size_t)BT * DM * 2);
  float*  gxf = (float*)alloc((size_t)BT * 3 * DM * 4);
  float*  Hsf = (float*)alloc((size_t)BT * DM * 4);
  __bf16* Hsb = (__bf16*)alloc((size_t)BT * DM * 2);
  float*  zf  = (float*)alloc((size_t)BT * DM * 4);
  __bf16* yb  = (__bf16*)alloc((size_t)BT * DM * 2);
  __bf16* qkvb = (__bf16*)alloc((size_t)BT * 3 * DM * 2);
  __bf16* attnb = (__bf16*)alloc((size_t)BT * DM * 2);
  __bf16* hid2b = (__bf16*)alloc((size_t)BT * 4 * DM * 2);
  __bf16* znb = (__bf16*)alloc((size_t)BT * DM * 2);
  __bf16* cqkvb = (__bf16*)alloc((size_t)BT * 3 * DM * 2);
  __bf16* cattnb = (__bf16*)alloc((size_t)BT * DM * 2);
  __bf16* aob = (__bf16*)alloc((size_t)BT * DM * 2);
  float*  prelnf = (float*)alloc((size_t)BT * DM * 4);

  long nMD = (long)MD * DM, nBT = (long)BT * DM;
  int bMD = (int)((nMD + 255) / 256), bBT = (int)((nBT + 255) / 256);

  // --- RK4 jump-ODE states ---
  k_delta<<<bMD, 256, 0, stream>>>(x, ts, deltaf, deltab);
  gemm(deltab, ode_w1b, ode_b1, nullptr, nullptr, hidb, MD, DM, DM, DM, ACT_TANH);
  gemm(hidb, ode_w2b, ode_b2, nullptr, f1, nullptr, MD, DM, DM, DM, ACT_NONE);
  k_axpy_stage<<<bMD, 256, 0, stream>>>(deltaf, f1, ts, 0.5f, stageb);
  gemm(stageb, ode_w1b, ode_b1, nullptr, nullptr, hidb, MD, DM, DM, DM, ACT_TANH);
  gemm(hidb, ode_w2b, ode_b2, nullptr, f2, nullptr, MD, DM, DM, DM, ACT_NONE);
  k_axpy_stage<<<bMD, 256, 0, stream>>>(deltaf, f2, ts, 0.5f, stageb);
  gemm(stageb, ode_w1b, ode_b1, nullptr, nullptr, hidb, MD, DM, DM, DM, ACT_TANH);
  gemm(hidb, ode_w2b, ode_b2, nullptr, f3, nullptr, MD, DM, DM, DM, ACT_NONE);
  k_axpy_stage<<<bMD, 256, 0, stream>>>(deltaf, f3, ts, 1.0f, stageb);
  gemm(stageb, ode_w1b, ode_b1, nullptr, nullptr, hidb, MD, DM, DM, DM, ACT_TANH);
  gemm(hidb, ode_w2b, ode_b2, nullptr, f4, nullptr, MD, DM, DM, DM, ACT_NONE);
  k_rk4_final<<<bBT, 256, 0, stream>>>(x, ts, f1, f2, f3, f4, Hf, Hb);

  // --- GRU: input proj then sequential scan ---
  gemm(Hb, wihb, gru_bih, nullptr, gxf, nullptr, BT, 3 * DM, DM, 3 * DM, ACT_NONE);
  k_gru_scan<<<1, 1024, 0, stream>>>(gxf, whhb, gru_bhh, Hsf, Hsb);

  // --- time encoding ---
  k_time_enc<<<bBT, 256, 0, stream>>>(x, ts, te_a, te_b, zf);

  // --- transformer encoder (pre-norm, causal) ---
  for (int l = 0; l < 2; ++l) {
    k_ln<<<BT, 256, 0, stream>>>(zf, t_ln1_w + l * DM, t_ln1_b + l * DM,
                                 nullptr, yb);
    gemm(yb, tinb + (size_t)l * 3 * DM * DM, t_in_b + l * 3 * DM, nullptr,
         nullptr, qkvb, BT, 3 * DM, DM, 3 * DM, ACT_NONE);
    k_flash<<<dim3(TT / 16 / 8, NH, BB), 256, 0, stream>>>(
        qkvb, 0, DM, 2 * DM, 3 * DM, attnb, DM, TT, 1);
    gemm(attnb, toutb + (size_t)l * DM * DM, t_out_b + l * DM, zf, zf, nullptr,
         BT, DM, DM, DM, ACT_NONE);
    k_ln<<<BT, 256, 0, stream>>>(zf, t_ln2_w + l * DM, t_ln2_b + l * DM,
                                 nullptr, yb);
    gemm(yb, ff1b + (size_t)l * 4 * DM * DM, t_ff1_b + l * 4 * DM, nullptr,
         nullptr, hid2b, BT, 4 * DM, DM, 4 * DM, ACT_RELU);
    gemm(hid2b, ff2b + (size_t)l * 4 * DM * DM, t_ff2_b + l * DM, zf, zf,
         nullptr, BT, DM, 4 * DM, DM, ACT_NONE);
  }
  k_ln<<<BT, 256, 0, stream>>>(zf, t_lnf_w, t_lnf_b, nullptr, znb);

  // --- cross attention: Q = GRU states, K/V = transformer output ---
  gemm(Hsb, cainb, ca_in_b, nullptr, nullptr, cqkvb, BT, DM, DM, 3 * DM,
       ACT_NONE);
  gemm(znb, cainb + (size_t)DM * DM, ca_in_b + DM, nullptr, nullptr,
       cqkvb + DM, BT, 2 * DM, DM, 3 * DM, ACT_NONE);
  k_flash<<<dim3(TT / 16 / 8, NH, BB), 256, 0, stream>>>(
      cqkvb, 0, DM, 2 * DM, 3 * DM, cattnb, DM, TT, 0);
  gemm(cattnb, caoutb, ca_out_b, nullptr, nullptr, aob, BT, DM, DM, DM,
       ACT_NONE);
  gemm(aob, calinb, ca_lin_b, Hsf, prelnf, nullptr, BT, DM, DM, DM, ACT_NONE);
  k_ln<<<BT, 256, 0, stream>>>(prelnf, ca_ln_w, ca_ln_b, (float*)d_out,
                               nullptr);
}
